// TransformerPredictor_im_68040871903599
// MI455X (gfx1250) — matrix operand order repaired and checked by reading, not compiled, not run on hardware
//
#include <hip/hip_runtime.h>
#include <hip/hip_bf16.h>

// CDNA5 / gfx1250 implementation.
// Heavy math runs through v_wmma_f32_16x16x32_bf16 (fp32 accumulate).
// GEMM2 (16384x4096x1024) is fused with the union gating and GEMM3
// (K=4096 -> 51) so the 16384x4096 visual_rep never touches HBM.
// The per-block 16x1024 bf16 prod tile is staged into LDS with the
// Tensor Data Mover (tensor_load_to_lds, TENSORcnt).

typedef __bf16 bf16;
typedef __attribute__((ext_vector_type(16))) __bf16        v16bf;
typedef __attribute__((ext_vector_type(8)))  float         v8f;
typedef __attribute__((ext_vector_type(4)))  unsigned int  u32x4;
typedef __attribute__((ext_vector_type(8)))  unsigned int  u32x8;

union FragCvt { u32x4 u[2]; v16bf v; };

__device__ __forceinline__ v8f wmma_bf16(v16bf a, v16bf b, v8f c) {
  // (neg_a, A, neg_b, B, c_mod, C, reuse_a, reuse_b)
  return __builtin_amdgcn_wmma_f32_16x16x32_bf16(false, a, false, b,
                                                 (short)0, c, false, false);
}

// Two 16-byte loads assembled into a 16xbf16 fragment register set.
__device__ __forceinline__ v16bf load_frag_2x16B(const bf16* p0, const bf16* p1) {
  FragCvt t;
  t.u[0] = *reinterpret_cast<const u32x4*>(p0);
  t.u[1] = *reinterpret_cast<const u32x4*>(p1);
  return t.v;
}

// A-matrix fragment (16x32 bf16, row-major source, ISA 7.12.2 layout):
// lane<16 : elems[0..7]=K k0+0..7,  elems[8..15]=K k0+16..23   (M = lane)
// lane>=16: elems[0..7]=K k0+8..15, elems[8..15]=K k0+24..31   (M = lane-16)
__device__ __forceinline__ v16bf load_a_frag(const bf16* A, int lda,
                                             int row0, int k0, int lane) {
  int m  = lane & 15;
  int kb = k0 + ((lane >> 4) << 3);
  const bf16* p = A + (size_t)(row0 + m) * lda + kb;
  return load_frag_2x16B(p, p + 16);
}

// B-matrix fragment (32x16 bf16) where B[k][n] = W[n][k], W row-major:
// lanes 0-15 : N = n0+lane,    elems[0..7] = K k0+0..7,  elems[8..15] = K k0+16..23
// lanes 16-31: N = n0+lane-16, elems[0..7] = K k0+8..15, elems[8..15] = K k0+24..31
__device__ __forceinline__ v16bf load_bT_frag(const bf16* W, int ldw,
                                              int n0, int k0, int lane) {
  int n  = n0 + (lane & 15);
  int kb = k0 + ((lane >> 4) << 3);
  const bf16* p = W + (size_t)n * ldw + kb;
  return load_frag_2x16B(p, p + 16);
}

// Issue a 2-D TDM copy: rows x rowbytes from global (row stride strideb)
// into contiguous LDS at lds_byte_off. data_size = 8B units.
// Descriptor layout per CDNA5 ISA 8.3/8.4 (groups 0 and 1, 2-D form).
__device__ __forceinline__ void tdm_load_2d(unsigned lds_byte_off,
                                            const void* gaddr,
                                            unsigned rowbytes, unsigned rows,
                                            unsigned strideb) {
  unsigned long long ga = (unsigned long long)(uintptr_t)gaddr;
  const unsigned u0 = rowbytes >> 3;   // tile_dim0 / tensor_dim0 in 8B units
  const unsigned su = strideb >> 3;    // tensor_dim0_stride in 8B units
  u32x4 g0;
  g0[0] = 1u;                                   // count=1, user descriptor
  g0[1] = lds_byte_off;                         // lds_addr
  g0[2] = (unsigned)(ga & 0xffffffffu);         // global_addr[31:0]
  g0[3] = (unsigned)((ga >> 32) & 0x01ffffffu)  // global_addr[56:32]
          | (2u << 30);                         // type = 2 ("image")
  u32x8 g1;
  g1[0] = 3u << 16;            // workgroup_mask=0, data_size=3 (8B)
  g1[1] = (u0 & 0xffffu) << 16;                 // tensor_dim0[15:0]
  g1[2] = ((u0 >> 16) & 0xffffu) | ((rows & 0xffffu) << 16); // dim0 hi | tensor_dim1 lo
  g1[3] = ((rows >> 16) & 0xffffu) | ((u0 & 0xffffu) << 16); // dim1 hi | tile_dim0
  g1[4] = rows & 0xffffu;                       // tile_dim1 (tile_dim2=0)
  g1[5] = su;                                   // tensor_dim0_stride[31:0]
  g1[6] = 0u;                                   // stride hi | tensor_dim1_stride lo
  g1[7] = 0u;
  asm volatile("tensor_load_to_lds %0, %1" :: "s"(g0), "s"(g1) : "memory");
}

// ---------------- helper conversion kernels ----------------

__global__ void cvt_bf16_kernel(const float* __restrict__ src,
                                bf16* __restrict__ dst, int n) {
  int i = blockIdx.x * blockDim.x + threadIdx.x;
  if (i < n) dst[i] = (bf16)src[i];
}

// convert + zero-pad trailing rows (for 51 -> 64 row padding of W_rel / W_ctx)
__global__ void cvt_pad_kernel(const float* __restrict__ src,
                               bf16* __restrict__ dst,
                               int rows_src, int rows_dst, int cols) {
  int i = blockIdx.x * blockDim.x + threadIdx.x;
  int total = rows_dst * cols;
  if (i < total) {
    int r = i / cols;
    dst[i] = (r < rows_src) ? (bf16)src[i] : (bf16)0.0f;
  }
}

// prod_rep gather: row r gets head(edge_rep[p0][0:512]) ++ tail(edge_rep[p1][512:1024])
__global__ void gather_prod_kernel(const float* __restrict__ edge_rep,
                                   const int* __restrict__ pair_idx,
                                   bf16* __restrict__ prod, int n_rel) {
  int i = blockIdx.x * blockDim.x + threadIdx.x;
  if (i >= n_rel * 1024) return;
  int r = i >> 10;
  int k = i & 1023;
  int src_row = (k < 512) ? pair_idx[2 * r] : pair_idx[2 * r + 1];
  prod[i] = (bf16)edge_rep[(size_t)src_row * 1024 + k];
}

// ---------------- GEMM1: edge_rep = edge_ctx @ W_post_emb.T + b ----------------
// grid (80, 16), block 128 (4 waves). Wave w owns N-tile blockIdx.y*64 + w*16.
__global__ void __launch_bounds__(128)
gemm1_kernel(const bf16* __restrict__ A,     // 1280 x 512
             const bf16* __restrict__ W,     // 1024 x 512 (row-major = B col-major)
             const float* __restrict__ bias, // 1024
             float* __restrict__ out) {      // 1280 x 1024
  const int lane = threadIdx.x & 31;
  const int wave = threadIdx.x >> 5;
  const int row0 = blockIdx.x * 16;
  const int n0   = blockIdx.y * 64 + wave * 16;
  v8f acc = {};
  for (int k = 0; k < 512; k += 32) {
    v16bf a = load_a_frag(A, 512, row0, k, lane);
    v16bf b = load_bT_frag(W, 512, n0, k, lane);
    acc = wmma_bf16(a, b, acc);
  }
  const int n = n0 + (lane & 15);
  const float bv = bias[n];
#pragma unroll
  for (int i = 0; i < 8; ++i) {
    int m = ((lane >> 4) << 3) + i;  // C/D layout: lanes>=16 hold M=8..15
    out[(size_t)(row0 + m) * 1024 + n] = acc[i] + bv;
  }
}

// ---------------- GEMM4 + biases + freq bias (writes d_out first) ----------------
// out[r][n] = prod[r] . W_ctx[n] + b_rel[n] + b_ctx[n] + freq[bias_idx[r]][n]
__global__ void __launch_bounds__(128)
ctx_out_kernel(const bf16* __restrict__ prod,   // 16384 x 1024
               const bf16* __restrict__ Wctx,   // 64 x 1024 (zero-padded)
               const float* __restrict__ b_rel, const float* __restrict__ b_ctx,
               const float* __restrict__ freq,  // 22801 x 51
               const int* __restrict__ pair_idx, const int* __restrict__ obj_preds,
               float* __restrict__ out) {       // 16384 x 51
  const int lane = threadIdx.x & 31;
  const int wave = threadIdx.x >> 5;
  const int row0 = blockIdx.x * 16;
  const int n0   = wave * 16;  // 0..48, padded to 64
  v8f acc = {};
  for (int k = 0; k < 1024; k += 32) {
    v16bf a = load_a_frag(prod, 1024, row0, k, lane);
    v16bf b = load_bT_frag(Wctx, 1024, n0, k, lane);
    acc = wmma_bf16(a, b, acc);
  }
  const int n = n0 + (lane & 15);
  if (n < 51) {
    const float bb = b_rel[n] + b_ctx[n];
#pragma unroll
    for (int i = 0; i < 8; ++i) {
      int r  = row0 + ((lane >> 4) << 3) + i;
      int o0 = obj_preds[pair_idx[2 * r]];
      int o1 = obj_preds[pair_idx[2 * r + 1]];
      int bi = o0 * 151 + o1;
      out[(size_t)r * 51 + n] = acc[i] + bb + freq[(size_t)bi * 51 + n];
    }
  }
}

// ---------------- fused GEMM2 -> gate*union -> GEMM3 (accumulates into d_out) ----
// block = 128 threads (4 waves), owns 16 rel rows. 64 chunks of 64 N-columns.
__global__ void __launch_bounds__(128)
fused_gate_rel_kernel(const bf16* __restrict__ prod,  // 16384 x 1024
                      const bf16* __restrict__ Wpc,   // 4096 x 1024
                      const float* __restrict__ b_pc, // 4096
                      const float* __restrict__ un,   // 16384 x 4096
                      const bf16* __restrict__ Wrel,  // 64 x 4096 (zero-padded)
                      float* __restrict__ out) {      // 16384 x 51 (+=)
  __shared__ bf16 a_lds[16 * 1024];  // staged prod tile: 32 KB
  __shared__ bf16 vis_lds[16 * 64];  // visual bf16 tile:  2 KB
  const int tid  = threadIdx.x;
  const int lane = tid & 31;
  const int wave = tid >> 5;
  const int row0 = blockIdx.x * 16;

  // Stage this block's 16x1024 bf16 prod tile into LDS via the Tensor Data
  // Mover: one 2-D DMA (16 rows x 2048 B, stride 2048 B), issued by wave 0.
  if (wave == 0) {
    tdm_load_2d((unsigned)(uintptr_t)&a_lds[0],
                prod + (size_t)row0 * 1024,
                /*rowbytes=*/2048u, /*rows=*/16u, /*strideb=*/2048u);
    __builtin_amdgcn_s_wait_tensorcnt(0);
  }
  __syncthreads();

  v8f acc3 = {};
  for (int nc = 0; nc < 64; ++nc) {
    const int nbase = nc * 64;
    const int n0    = nbase + wave * 16;

    // GEMM2: ctx_gate tile (16 x 16 per wave), K = 1024
    v8f acc2 = {};
#pragma unroll 4
    for (int k = 0; k < 1024; k += 32) {
      v16bf a = load_a_frag(a_lds, 1024, 0, k, lane);
      v16bf b = load_bT_frag(Wpc, 1024, n0, k, lane);
      acc2 = wmma_bf16(a, b, acc2);
    }

    __syncthreads();  // previous iteration's GEMM3 reads of vis_lds are done
    {
      const int n   = n0 + (lane & 15);
      const float bpc = b_pc[n];
#pragma unroll
      for (int i = 0; i < 8; ++i) {
        int m = ((lane >> 4) << 3) + i;
        float u = un[(size_t)(row0 + m) * 4096 + n];
        float v = (acc2[i] + bpc) * u;                     // visual_rep element
        vis_lds[m * 64 + wave * 16 + (lane & 15)] = (bf16)v;
      }
      if (nc + 1 < 64)  // warm next union chunk (global_prefetch_b8)
        __builtin_prefetch(&un[(size_t)row0 * 4096 + nbase + 64], 0, 1);
    }
    __syncthreads();  // visual tile visible to all waves

    // GEMM3 partial: acc3 += visual(16x64) @ Wrel_chunk.T ; wave owns N-tile wave*16
#pragma unroll
    for (int kk = 0; kk < 2; ++kk) {
      v16bf a3 = load_a_frag(vis_lds, 64, 0, kk * 32, lane);
      v16bf b3 = load_bT_frag(Wrel, 4096, wave * 16, nbase + kk * 32, lane);
      acc3 = wmma_bf16(a3, b3, acc3);
    }
  }

  const int n = wave * 16 + (lane & 15);
  if (n < 51) {
#pragma unroll
    for (int i = 0; i < 8; ++i) {
      int r = row0 + ((lane >> 4) << 3) + i;
      out[(size_t)r * 51 + n] += acc3[i];   // ctx_out_kernel ran first on stream
    }
  }
}

// ---------------- host side ----------------

extern "C" void kernel_launch(void* const* d_in, const int* in_sizes, int n_in,
                              void* d_out, int out_size, void* d_ws, size_t ws_size,
                              hipStream_t stream) {
  const float* edge_ctx  = (const float*)d_in[0];   // 1280 x 512
  const int*   pair_idx  = (const int*)d_in[1];     // 16384 x 2
  const int*   obj_preds = (const int*)d_in[2];     // 1280
  const float* un        = (const float*)d_in[3];   // 16384 x 4096
  const float* W_pe      = (const float*)d_in[4];   // 1024 x 512
  const float* b_pe      = (const float*)d_in[5];   // 1024
  const float* W_pc      = (const float*)d_in[6];   // 4096 x 1024
  const float* b_pc      = (const float*)d_in[7];   // 4096
  const float* W_rel     = (const float*)d_in[8];   // 51 x 4096
  const float* b_rel     = (const float*)d_in[9];   // 51
  const float* W_ctx     = (const float*)d_in[10];  // 51 x 1024
  const float* b_ctx     = (const float*)d_in[11];  // 51
  const float* freq      = (const float*)d_in[12];  // 22801 x 51
  float* out = (float*)d_out;                       // 16384 x 51

  // workspace carve-out (~50 MB total), 256B aligned slabs
  char* ws = (char*)d_ws;
  size_t off = 0;
  auto alloc = [&](size_t bytes) -> void* {
    void* p = ws + off;
    off = (off + bytes + 255) & ~(size_t)255;
    return p;
  };
  bf16*  ec_bf    = (bf16*)alloc(1280ull * 512 * 2);
  bf16*  wpe_bf   = (bf16*)alloc(1024ull * 512 * 2);
  bf16*  wpc_bf   = (bf16*)alloc(4096ull * 1024 * 2);
  bf16*  wctx_bf  = (bf16*)alloc(64ull * 1024 * 2);
  bf16*  wrel_bf  = (bf16*)alloc(64ull * 4096 * 2);
  float* edge_rep = (float*)alloc(1280ull * 1024 * 4);
  bf16*  prod_bf  = (bf16*)alloc(16384ull * 1024 * 2);

  const int T = 256;
  cvt_bf16_kernel<<<(1280 * 512 + T - 1) / T, T, 0, stream>>>(edge_ctx, ec_bf, 1280 * 512);
  cvt_bf16_kernel<<<(1024 * 512 + T - 1) / T, T, 0, stream>>>(W_pe, wpe_bf, 1024 * 512);
  cvt_bf16_kernel<<<(4096 * 1024 + T - 1) / T, T, 0, stream>>>(W_pc, wpc_bf, 4096 * 1024);
  cvt_pad_kernel<<<(64 * 1024 + T - 1) / T, T, 0, stream>>>(W_ctx, wctx_bf, 51, 64, 1024);
  cvt_pad_kernel<<<(64 * 4096 + T - 1) / T, T, 0, stream>>>(W_rel, wrel_bf, 51, 64, 4096);

  gemm1_kernel<<<dim3(80, 16), 128, 0, stream>>>(ec_bf, wpe_bf, b_pe, edge_rep);

  gather_prod_kernel<<<(16384 * 1024) / T, T, 0, stream>>>(edge_rep, pair_idx, prod_bf, 16384);

  // writes every output element (d_out is poisoned before timing)
  ctx_out_kernel<<<1024, 128, 0, stream>>>(prod_bf, wctx_bf, b_rel, b_ctx, freq,
                                           pair_idx, obj_preds, out);
  // accumulates the visual path on top
  fused_gate_rel_kernel<<<1024, 128, 0, stream>>>(prod_bf, wpc_bf, b_pc, un, wrel_bf, out);
}
